// QuantumVelocityField_14568529068579
// MI455X (gfx1250) — matrix operand order repaired and checked by reading, not compiled, not run on hardware
//
#include <hip/hip_runtime.h>
#include <math.h>

typedef __attribute__((ext_vector_type(16))) _Float16 v16h;
typedef __attribute__((ext_vector_type(8)))  float    v8f;

#ifndef __has_builtin
#define __has_builtin(x) 0
#endif
#if __has_builtin(__builtin_amdgcn_tensor_load_to_lds)
#define HAVE_TDM 1
#else
#define HAVE_TDM 0
#endif

static constexpr int NQ      = 10;
static constexpr int DIM     = 1 << NQ;   // 1024
static constexpr int LAT     = 128;
static constexpr int NB      = 4;
static constexpr int VD      = 6;
static constexpr int NOBS    = 9;
static constexpr int ENC_DIM = NB * NQ;   // 40

// Pre-shuffled f16 weights in workspace, stored in WMMA B-fragment order:
//   Ws[tile][kk][lane][i],  k = kk*32 + 16*(i>>3) + 8*(lane>>4) + (i&7),  n = tile*16 + (lane&15)
// Out-of-range (K,N) positions are zero-filled, so hot loops need no guards.
static constexpr int W1S_N = 16 * 8 * 512;   // 256x256            = 65536
static constexpr int W2S_N = 3  * 8 * 512;   // 256x40  (N pad 48) = 12288
static constexpr int W3S_N = 16 * 1 * 512;   //   9x256 (K pad 32) =  8192
static constexpr int W4S_N = 8  * 8 * 512;   // 256x128            = 32768

__device__ inline float silu_f(float v) { return v / (1.0f + __expf(-v)); }

// ---------------------------------------------------------------------------
// A-operand swizzle (CDNA5 ISA 7.12.2, 16-bit A 16x32): a lane's 16 halves are
// contiguous & 32B aligned:  sA[kk*512 + lane*16 + i],
//   k = kk*32 + 16*(i>>3) + 8*(lane>>4) + (i&7),  m = lane&15
// ---------------------------------------------------------------------------
__device__ inline int a_swz(int m, int k) {
  const int kk = k >> 5, kr = k & 31;
  const int g  = (kr >> 3) & 1;
  const int i  = ((kr >> 4) << 3) | (kr & 7);
  return kk * 512 + (((g << 4) | m) << 4) + i;
}

// ---------------------------------------------------------------------------
// Tensor Data Mover: contiguous / strided 2D f16 tile, global -> LDS.
// D# per CDNA5 ISA.
// ---------------------------------------------------------------------------
#if HAVE_TDM
typedef unsigned int u32x4 __attribute__((ext_vector_type(4)));
typedef int          i32x4 __attribute__((ext_vector_type(4)));
typedef int          i32x8 __attribute__((ext_vector_type(8)));

__device__ inline void tdm_load_2d_f16(const _Float16* gsrc, unsigned lds_off,
                                       unsigned tile_n, unsigned tile_k, unsigned ld) {
  const unsigned long long ga = (unsigned long long)(uintptr_t)gsrc;
  u32x4 g0;
  g0[0] = 1u;                                               // count=1 (valid user D#)
  g0[1] = lds_off;                                          // lds_addr
  g0[2] = (unsigned)(ga & 0xFFFFFFFFu);                     // global_addr[31:0]
  g0[3] = (unsigned)((ga >> 32) & 0x1FFFFFFu) | (2u << 30); // [56:32] | type=2
  i32x8 g1;
  g1[0] = (int)(1u << 16);                                  // data_size=1 (2 bytes)
  g1[1] = (int)((ld & 0xFFFFu) << 16);                      // tensor_dim0 lo16
  g1[2] = (int)((ld >> 16) | ((tile_k & 0xFFFFu) << 16));   // dim0 hi | tensor_dim1 lo
  g1[3] = (int)((tile_k >> 16) | (tile_n << 16));           // dim1 hi | tile_dim0
  g1[4] = (int)(tile_k & 0xFFFFu);                          // tile_dim1 (tile_dim2=0)
  g1[5] = (int)ld;                                          // tensor_dim0_stride lo32
  g1[6] = 0;
  g1[7] = 0;
  i32x4 z4 = {0, 0, 0, 0};
#if defined(__clang_major__) && (__clang_major__ >= 23)
  i32x8 z8 = {0, 0, 0, 0, 0, 0, 0, 0};
  __builtin_amdgcn_tensor_load_to_lds(g0, g1, z4, z4, z8, 0);
#else
  __builtin_amdgcn_tensor_load_to_lds(g0, g1, z4, z4, 0);
#endif
}

__device__ inline void wait_tensor0() {
#if __has_builtin(__builtin_amdgcn_s_wait_tensorcnt)
  __builtin_amdgcn_s_wait_tensorcnt(0);
#else
  asm volatile("s_wait_tensorcnt 0x0" ::: "memory");
#endif
}
#endif // HAVE_TDM

// ---------------------------------------------------------------------------
// Kernel 0: f32 -> f16 weight conversion + B-fragment shuffle (one time; the
// four shuffled matrices total 232 KB f16 -> L2-resident for the whole run).
// ---------------------------------------------------------------------------
__device__ inline void shuffle_one(const float* __restrict__ W, _Float16* __restrict__ Ws,
                                   int K, int N, int chunks, int idx) {
  const int perTile = chunks * 512;
  const int tile = idx / perTile;
  const int rem  = idx % perTile;
  const int kk   = rem >> 9;
  const int r2   = rem & 511;
  const int lane = r2 >> 4;
  const int i    = r2 & 15;
  const int k    = kk * 32 + ((i >> 3) << 4) + ((lane >> 4) << 3) + (i & 7);
  const int n    = tile * 16 + (lane & 15);
  Ws[idx] = (k < K && n < N) ? (_Float16)W[k * N + n] : (_Float16)0.0f;
}

__global__ __launch_bounds__(256)
void qvf_prep(const float* __restrict__ W1, const float* __restrict__ W2,
              const float* __restrict__ W3, const float* __restrict__ W4,
              _Float16* __restrict__ W1s, _Float16* __restrict__ W2s,
              _Float16* __restrict__ W3s, _Float16* __restrict__ W4s) {
  const int i = blockIdx.x * 256 + threadIdx.x;
  if (i < W1S_N) shuffle_one(W1, W1s, 256, 256,     8, i);
  if (i < W2S_N) shuffle_one(W2, W2s, 256, ENC_DIM, 8, i);
  if (i < W3S_N) shuffle_one(W3, W3s, NOBS, 256,    1, i);
  if (i < W4S_N) shuffle_one(W4, W4s, 256, LAT,     8, i);
}

// ---------------------------------------------------------------------------
// Kernel 1: x = [z_t | cos | sin]; h = silu(x@W1+b1); enc = h@W2+b2
// One block = 16 batch rows, 128 threads (4 waves). Each wave TDM-pulls its
// pre-shuffled 8 KB W1 panel into LDS (contiguous 1-D tile), then every WMMA
// operand (A and B) is a single 32B vector LDS load.
// ---------------------------------------------------------------------------
__global__ __launch_bounds__(128)
void qvf_encoder(const float* __restrict__ z_t, const float* __restrict__ t,
                 const _Float16* __restrict__ W1s, const float* __restrict__ b1,
                 const _Float16* __restrict__ W2s, const float* __restrict__ b2,
                 float* __restrict__ enc) {
  __shared__ __align__(32) _Float16 sX[4096];      // 16x256 A, fragment-swizzled
  __shared__ __align__(32) _Float16 sH[4096];      // 16x256 A, fragment-swizzled
  __shared__ __align__(32) _Float16 sB[4][4096];   // per-wave pre-shuffled B panel
  const int tid  = threadIdx.x;
  const int lane = tid & 31;
  const int wave = tid >> 5;
  const int b0   = blockIdx.x * 16;

  __builtin_prefetch(&W1s[tid * 64], 0, 1);        // global_prefetch_b8 -> L2

  // Build the f16 input tile in swizzled A layout: [z_t(128) | cos(64) | sin(64)]
  for (int e = tid; e < 4096; e += 128) {
    int m = e >> 8, c = e & 255;
    float v;
    if (c < 128) {
      v = z_t[(b0 + m) * LAT + c];
    } else {
      int j     = (c < 192) ? (c - 128) : (c - 192);
      float f   = __expf(-9.210340372f * (float)j * (1.0f / 64.0f)); // exp(-ln 1e4 * j/64)
      float arg = t[b0 + m] * f;
      v = (c < 192) ? __cosf(arg) : __sinf(arg);
    }
    sX[a_swz(m, c)] = (_Float16)v;
  }
  __syncthreads();

  // h = silu(x @ W1 + b1): 16 col tiles, 4 per wave; K = 256 -> 8 WMMAs each
  for (int it = 0; it < 4; ++it) {
    const int tile = it * 4 + wave;
    const _Float16* panel = W1s + tile * 4096;
#if HAVE_TDM
    tdm_load_2d_f16(panel, (unsigned)(uintptr_t)&sB[wave][0], 4096u, 1u, 4096u);
    wait_tensor0();
#else
    for (int r = lane; r < 256; r += 32)
      *(v16h*)&sB[wave][r * 16] = *(const v16h*)(panel + r * 16);
#endif
    __syncthreads();

    v8f acc = {};
#pragma unroll
    for (int kk = 0; kk < 8; ++kk) {
      v16h a  = *(const v16h*)&sX[kk * 512 + lane * 16];        // 2x ds_load_b128
      v16h bb = *(const v16h*)&sB[wave][kk * 512 + lane * 16];  // 2x ds_load_b128
      acc = __builtin_amdgcn_wmma_f32_16x16x32_f16(false, a, false, bb,
                                                   (short)0, acc, false, false);
    }
    const int n = lane & 15, g = lane >> 4;
    const float bias = b1[tile * 16 + n];
#pragma unroll
    for (int r = 0; r < 8; ++r)
      sH[a_swz(r + (g << 3), tile * 16 + n)] = (_Float16)silu_f(acc[r] + bias);
  }
  __syncthreads();

  // enc = h @ W2 + b2: 40 cols -> 3 col tiles on waves 0..2 (B direct from L2)
  if (wave < 3) {
    v8f acc = {};
#pragma unroll
    for (int kk = 0; kk < 8; ++kk) {
      v16h a  = *(const v16h*)&sH[kk * 512 + lane * 16];
      v16h bb = *(const v16h*)&W2s[(wave * 8 + kk) * 512 + lane * 16];
      acc = __builtin_amdgcn_wmma_f32_16x16x32_f16(false, a, false, bb,
                                                   (short)0, acc, false, false);
    }
    const int n = lane & 15, g = lane >> 4;
    const int col = wave * 16 + n;
    if (col < ENC_DIM) {
      const float bias = b2[col];
#pragma unroll
      for (int r = 0; r < 8; ++r)
        enc[(b0 + r + (g << 3)) * ENC_DIM + col] = acc[r] + bias;
    }
  }
}

// ---------------------------------------------------------------------------
// Kernel 2: 10-qubit statevector sim, entirely LDS-resident (8 KB / state).
// One block (256 threads, 8 waves) per batch row. RY: 512 amp pairs, 2/thread.
// CNOT: 256 swaps, 1/thread. Then 9 two-qubit expectation values.
// ---------------------------------------------------------------------------
__global__ __launch_bounds__(256)
void qvf_circuit(const float* __restrict__ enc, const float* __restrict__ var_params,
                 const float* __restrict__ A_p, const float* __restrict__ B_p,
                 const float* __restrict__ D_p, float* __restrict__ obs) {
  __shared__ float sRe[DIM];
  __shared__ float sIm[DIM];
  __shared__ float sObs[NOBS];
  const int tid = threadIdx.x;
  const int b   = blockIdx.x;

  for (int i = tid; i < DIM; i += 256) {
    sRe[i] = (i == 0) ? 1.0f : 0.0f;
    sIm[i] = 0.0f;
  }
  if (tid < NOBS) sObs[tid] = 0.0f;
  __syncthreads();

  auto cnot = [&](int ctrl) {
    // qubit q <-> bit (9-q); tgt = ctrl+1 -> target bit bt, control bit bt+1
    const int bt = 8 - ctrl;
    const int lo = tid & ((1 << bt) - 1);
    const int hi = tid >> bt;
    const int i0 = (hi << (bt + 2)) | (1 << (bt + 1)) | lo;
    const int i1 = i0 | (1 << bt);
    float tr = sRe[i0]; sRe[i0] = sRe[i1]; sRe[i1] = tr;
    float ti = sIm[i0]; sIm[i0] = sIm[i1]; sIm[i1] = ti;
    __syncthreads();
  };

  for (int layer = 0; layer < NB + VD; ++layer) {
    const bool encLayer = layer < NB;
    for (int q = 0; q < NQ; ++q) {
      const float th = encLayer ? enc[b * ENC_DIM + layer * NQ + q]
                                : var_params[(layer - NB) * NQ + q];
      const float ch = __cosf(0.5f * th);
      const float sh = __sinf(0.5f * th);
      const int sb = NQ - 1 - q;
      const int stride = 1 << sb;
#pragma unroll
      for (int pp = 0; pp < 2; ++pp) {
        const int p  = tid + pp * 256;               // disjoint pair ids 0..511
        const int lo = p & (stride - 1);
        const int i0 = ((p >> sb) << (sb + 1)) | lo;
        const int i1 = i0 | stride;
        float r0 = sRe[i0], r1 = sRe[i1];
        float m0 = sIm[i0], m1 = sIm[i1];
        sRe[i0] = ch * r0 - sh * r1;  sRe[i1] = sh * r0 + ch * r1;
        sIm[i0] = ch * m0 - sh * m1;  sIm[i1] = sh * m0 + ch * m1;
      }
      __syncthreads();
    }
    if (layer != NB - 1) {
      for (int q = 0; q < NQ - 1; q += 2) cnot(q);
      for (int q = 1; q < NQ - 1; q += 2) cnot(q);
    }
  }

  // <psi| H_s |psi> for the 9 two-qubit observables on qubits (s, s+1)
  for (int s = 0; s < NOBS; ++s) {
    const int shr  = 8 - s;
    const int a    = tid >> shr;
    const int r    = tid & ((1 << shr) - 1);
    const int base = (a << (shr + 2)) | r;
    float xs[4], ys[4];
#pragma unroll
    for (int l = 0; l < 4; ++l) { xs[l] = sRe[base + (l << shr)]; ys[l] = sIm[base + (l << shr)]; }

    float c = 2.0f * D_p[s * 4 + 1] * (xs[0] * xs[0] + ys[0] * ys[0])
            + 2.0f * D_p[s * 4 + 2] * (xs[1] * xs[1] + ys[1] * ys[1])
            + 2.0f * D_p[s * 4 + 3] * (xs[2] * xs[2] + ys[2] * ys[2]);
    const int lidx[6] = {1, 2, 2, 3, 3, 3};
    const int kidx[6] = {0, 0, 1, 0, 1, 2};
#pragma unroll
    for (int j = 0; j < 6; ++j) {
      const float Av = A_p[s * 6 + j], Bv = B_p[s * 6 + j];
      const float xl = xs[lidx[j]], yl = ys[lidx[j]];
      const float xk = xs[kidx[j]], yk = ys[kidx[j]];
      c += 2.0f * (Av * (xl * xk + yl * yk) - Bv * (xl * yk - yl * xk));
    }
#pragma unroll
    for (int off = 16; off > 0; off >>= 1) c += __shfl_xor(c, off, 32);
    if ((tid & 31) == 0) atomicAdd(&sObs[s], c);
  }
  __syncthreads();
  if (tid < NOBS) obs[b * NOBS + tid] = sObs[tid];
}

// ---------------------------------------------------------------------------
// Kernel 3: out = silu(obs @ W3 + b3) @ W4 + b4
// One block = 16 batch rows, 128 threads (4 waves). K=9 zero-padded to one
// WMMA; all B fragments are direct 32B vector loads from pre-shuffled L2 data.
// ---------------------------------------------------------------------------
__global__ __launch_bounds__(128)
void qvf_decoder(const float* __restrict__ obs,
                 const _Float16* __restrict__ W3s, const float* __restrict__ b3,
                 const _Float16* __restrict__ W4s, const float* __restrict__ b4,
                 float* __restrict__ out) {
  __shared__ __align__(32) _Float16 sA[512];       // 16x32 (K padded), swizzled
  __shared__ __align__(32) _Float16 sY[4096];      // 16x256, swizzled
  const int tid  = threadIdx.x;
  const int lane = tid & 31;
  const int wave = tid >> 5;
  const int b0   = blockIdx.x * 16;

  for (int e = tid; e < 512; e += 128) {
    int m = e >> 5, k = e & 31;
    sA[a_swz(m, k)] = (_Float16)((k < NOBS) ? obs[(b0 + m) * NOBS + k] : 0.0f);
  }
  __syncthreads();

  // y = silu(obs @ W3 + b3): 16 col tiles / 4 waves, one WMMA each
  for (int tIdx = wave; tIdx < 16; tIdx += 4) {
    v16h a  = *(const v16h*)&sA[lane * 16];
    v16h bb = *(const v16h*)&W3s[tIdx * 512 + lane * 16];
    v8f acc = {};
    acc = __builtin_amdgcn_wmma_f32_16x16x32_f16(false, a, false, bb,
                                                 (short)0, acc, false, false);
    const int n = lane & 15, g = lane >> 4;
    const float bias = b3[tIdx * 16 + n];
#pragma unroll
    for (int r = 0; r < 8; ++r)
      sY[a_swz(r + (g << 3), tIdx * 16 + n)] = (_Float16)silu_f(acc[r] + bias);
  }
  __syncthreads();

  // out = y @ W4 + b4: 8 col tiles / 4 waves; K=256 -> 8 WMMAs each
  for (int tIdx = wave; tIdx < 8; tIdx += 4) {
    v8f acc = {};
#pragma unroll
    for (int kk = 0; kk < 8; ++kk) {
      v16h a  = *(const v16h*)&sY[kk * 512 + lane * 16];
      v16h bb = *(const v16h*)&W4s[(tIdx * 8 + kk) * 512 + lane * 16];
      acc = __builtin_amdgcn_wmma_f32_16x16x32_f16(false, a, false, bb,
                                                   (short)0, acc, false, false);
    }
    const int n = lane & 15, g = lane >> 4;
    const float bias = b4[tIdx * 16 + n];
#pragma unroll
    for (int r = 0; r < 8; ++r)
      out[(b0 + r + (g << 3)) * LAT + tIdx * 16 + n] = acc[r] + bias;
  }
}

// ---------------------------------------------------------------------------
extern "C" void kernel_launch(void* const* d_in, const int* in_sizes, int n_in,
                              void* d_out, int out_size, void* d_ws, size_t ws_size,
                              hipStream_t stream) {
  const float* z_t = (const float*)d_in[0];
  const float* t   = (const float*)d_in[1];
  const float* W1  = (const float*)d_in[2];
  const float* b1  = (const float*)d_in[3];
  const float* W2  = (const float*)d_in[4];
  const float* b2  = (const float*)d_in[5];
  const float* var = (const float*)d_in[6];
  const float* A_p = (const float*)d_in[7];
  const float* B_p = (const float*)d_in[8];
  const float* D_p = (const float*)d_in[9];
  const float* W3  = (const float*)d_in[10];
  const float* b3  = (const float*)d_in[11];
  const float* W4  = (const float*)d_in[12];
  const float* b4  = (const float*)d_in[13];
  float* out = (float*)d_out;

  const int Bn = in_sizes[1];                        // batch (2048)

  // Workspace: [enc f32 | obs f32 | W1s..W4s f16, fragment-shuffled]
  float* enc = (float*)d_ws;                         // Bn x 40
  float* obs = enc + (size_t)Bn * ENC_DIM;           // Bn x 9
  _Float16* W1s = (_Float16*)(obs + (size_t)Bn * NOBS);  // 32B-aligned
  _Float16* W2s = W1s + W1S_N;
  _Float16* W3s = W2s + W2S_N;
  _Float16* W4s = W3s + W3S_N;

  qvf_prep   <<<(W1S_N + 255) / 256, 256, 0, stream>>>(W1, W2, W3, W4, W1s, W2s, W3s, W4s);
  qvf_encoder<<<Bn / 16, 128, 0, stream>>>(z_t, t, W1s, b1, W2s, b2, enc);
  qvf_circuit<<<Bn, 256, 0, stream>>>(enc, var, A_p, B_p, D_p, obs);
  qvf_decoder<<<Bn / 16, 128, 0, stream>>>(obs, W3s, b3, W4s, b4, out);
}
